// GAT_18640158065247
// MI455X (gfx1250) — matrix operand order repaired and checked by reading, not compiled, not run on hardware
//
#include <hip/hip_runtime.h>
#include <hip/hip_bf16.h>
#include <math.h>

// ---------------- problem constants (match reference) ----------------
#define NN      50000
#define EE      800000
#define ETOT    (EE + NN)        // self-loops appended
#define NFEAT   128
#define NHID    16
#define HEADS   8
#define HID     (HEADS * NHID)   // 128
#define NCLASS  2
#define NEG_SLOPE 0.2f
#define NTILES  (NN / 16)        // 3125, exact
#define WLDS_STRIDE 68           // words per column (64 data + 4 pad); 272 B, 16B-aligned

typedef __attribute__((ext_vector_type(16))) _Float16 v16h;
typedef __attribute__((ext_vector_type(8)))  float    v8f;

// ------------- helpers -------------
__device__ __forceinline__ void atomicMaxF(float* addr, float val) {
  // ordered-float atomic max via int/uint punning; valid for non-NaN values,
  // init value -inf.
  if (val >= 0.0f) atomicMax((int*)addr, __float_as_int(val));
  else             atomicMin((unsigned int*)addr, __float_as_uint(val));
}

__global__ __launch_bounds__(256) void zero_kernel(float* __restrict__ p, long n) {
  long i = (long)blockIdx.x * blockDim.x + threadIdx.x;
  long stride = (long)gridDim.x * blockDim.x;
  for (; i < n; i += stride) p[i] = 0.0f;
}

// ------------- layer-1 GEMM: h1 = x @ W1 via WMMA f32<=f16 -------------
// Block = 8 waves. W1 is staged once per block into LDS, transposed and
// converted to f16 (column-major: 16 contiguous K-halves per B fragment).
// Each wave then computes a 16-row tile: 8 col-tiles x 4 k-steps of 32.
__global__ __launch_bounds__(256) void gemm1_wmma(const float* __restrict__ X,
                                                  const float* __restrict__ W,
                                                  float* __restrict__ H) {
  __shared__ __align__(16) unsigned int wlds[HID * WLDS_STRIDE]; // ~34 KB

  // ---- stage W1^T as packed f16 pairs: wlds[col*STRIDE + kp] = {K=2kp, K=2kp+1}
  for (int idx = threadIdx.x; idx < HID * (NFEAT / 2); idx += 256) {
    int col = idx & (HID - 1);
    int kp  = idx >> 7;
    float w0 = W[(size_t)(2 * kp)     * HID + col];
    float w1 = W[(size_t)(2 * kp + 1) * HID + col];
    union { _Float16 h[2]; unsigned int u; } pk;
    pk.h[0] = (_Float16)w0;
    pk.h[1] = (_Float16)w1;
    wlds[col * WLDS_STRIDE + kp] = pk.u;
  }
  __syncthreads();   // all waves reach the barrier before any tile early-exit

  const int wave = threadIdx.x >> 5;
  const int lane = threadIdx.x & 31;
  const int tile = blockIdx.x * 8 + wave;
  if (tile >= NTILES) return;               // uniform per wave -> EXEC all-1s below

  const int lrow = lane & 15;
  const int row  = tile * 16 + lrow;        // A: M = lane%16
  const int aklo = (lane < 16) ? 0 : 16;    // A K-offset per half-wave (floats)
  const int bkp  = (lane < 16) ? 0 : 8;     // B K-pair offset per half-wave (words)

  // A fragments for all 4 k-steps (contiguous k = 16h + s per half-wave, same as B), float4 loads
  v16h afrag[4];
#pragma unroll
  for (int ks = 0; ks < 4; ++ks) {
    const float4* xp = (const float4*)(X + (size_t)row * NFEAT + ks * 32 + aklo);
    float4 x0 = xp[0], x1 = xp[1];          // K = klo + 0..7
    float4 x2 = xp[2], x3 = xp[3];          // K = klo + 8..15
    afrag[ks][0]  = (_Float16)x0.x; afrag[ks][1]  = (_Float16)x0.y;
    afrag[ks][2]  = (_Float16)x0.z; afrag[ks][3]  = (_Float16)x0.w;
    afrag[ks][4]  = (_Float16)x1.x; afrag[ks][5]  = (_Float16)x1.y;
    afrag[ks][6]  = (_Float16)x1.z; afrag[ks][7]  = (_Float16)x1.w;
    afrag[ks][8]  = (_Float16)x2.x; afrag[ks][9]  = (_Float16)x2.y;
    afrag[ks][10] = (_Float16)x2.z; afrag[ks][11] = (_Float16)x2.w;
    afrag[ks][12] = (_Float16)x3.x; afrag[ks][13] = (_Float16)x3.y;
    afrag[ks][14] = (_Float16)x3.z; afrag[ks][15] = (_Float16)x3.w;
  }

#pragma unroll
  for (int nt = 0; nt < 8; ++nt) {
    const int col = nt * 16 + lrow;          // B/C: N = lane%16
    v8f c = {};
#pragma unroll
    for (int ks = 0; ks < 4; ++ks) {
      // lane needs halves K = ks*32 + bkp*2 + 0..15 of column `col`:
      // 8 contiguous LDS words, 16B aligned -> two ds_load_b128
      const uint4* bp = (const uint4*)&wlds[col * WLDS_STRIDE + ks * 16 + bkp];
      union { uint4 q[2]; v16h v; } bb;
      bb.q[0] = bp[0];
      bb.q[1] = bp[1];
      c = __builtin_amdgcn_wmma_f32_16x16x32_f16(false, afrag[ks],
                                                 false, bb.v,
                                                 (short)0, c, false, false);
    }
    // C layout: VGPR j -> row (j + 8*(lane>=16)), col lane%16
    const int mrow = tile * 16 + ((lane < 16) ? 0 : 8);
    float* hp = H + (size_t)mrow * HID + col;
#pragma unroll
    for (int j = 0; j < 8; ++j) hp[(size_t)j * HID] = c[j];
  }
}

// ------------- per-node attention coefficients, layer 1 -------------
__global__ __launch_bounds__(256) void alpha1_kernel(const float* __restrict__ H,
                                                     const float* __restrict__ attS,
                                                     const float* __restrict__ attD,
                                                     float* __restrict__ aS,
                                                     float* __restrict__ aD,
                                                     float* __restrict__ m1,
                                                     float* __restrict__ dn1) {
  int idx = blockIdx.x * blockDim.x + threadIdx.x;
  if (idx >= NN * HEADS) return;
  int n  = idx >> 3;
  int hd = idx & 7;
  const float* hp = H + (size_t)n * HID + hd * NHID;
  const float* s  = attS + hd * NHID;
  const float* d  = attD + hd * NHID;
  float ss = 0.f, dd = 0.f;
#pragma unroll
  for (int i = 0; i < NHID; ++i) { float h = hp[i]; ss += h * s[i]; dd += h * d[i]; }
  aS[idx] = ss; aD[idx] = dd;
  m1[idx] = -INFINITY; dn1[idx] = 0.0f;
}

__device__ __forceinline__ void edge_nodes(const int* __restrict__ ei, long e, int& s, int& d) {
  if (e < EE) { s = ei[e]; d = ei[EE + e]; }
  else        { s = d = (int)(e - EE); }
}

// ------------- layer-1 edge passes -------------
__global__ __launch_bounds__(256) void edge_a1(const int* __restrict__ ei,
                                               const float* __restrict__ aS,
                                               const float* __restrict__ aD,
                                               float* __restrict__ ebuf,
                                               float* __restrict__ m1) {
  long idx = (long)blockIdx.x * blockDim.x + threadIdx.x;
  if (idx >= (long)ETOT * HEADS) return;
  long e = idx >> 3; int hd = (int)(idx & 7);
  int s, d; edge_nodes(ei, e, s, d);
  float v = aS[(size_t)s * HEADS + hd] + aD[(size_t)d * HEADS + hd];
  v = (v < 0.f) ? NEG_SLOPE * v : v;
  ebuf[idx] = v;
  atomicMaxF(&m1[(size_t)d * HEADS + hd], v);
}

__global__ __launch_bounds__(256) void edge_b1(const int* __restrict__ ei,
                                               const float* __restrict__ m1,
                                               float* __restrict__ ebuf,
                                               float* __restrict__ dn1) {
  long idx = (long)blockIdx.x * blockDim.x + threadIdx.x;
  if (idx >= (long)ETOT * HEADS) return;
  long e = idx >> 3; int hd = (int)(idx & 7);
  int s, d; edge_nodes(ei, e, s, d);
  float ex = __expf(ebuf[idx] - m1[(size_t)d * HEADS + hd]);
  ebuf[idx] = ex;
  atomicAdd(&dn1[(size_t)d * HEADS + hd], ex);
}

// one wave per edge: 32 lanes x float4 = 128 features; head = lane/4
__global__ __launch_bounds__(256) void edge_c1(const int* __restrict__ ei,
                                               const float* __restrict__ ebuf,
                                               const float* __restrict__ dn1,
                                               const float* __restrict__ H,
                                               float* __restrict__ agg) {
  int wave = threadIdx.x >> 5;
  int lane = threadIdx.x & 31;
  long e = (long)blockIdx.x * 8 + wave;
  if (e >= ETOT) return;
  int s, d; edge_nodes(ei, e, s, d);
  int hd = lane >> 2;
  float a = ebuf[e * HEADS + hd] / dn1[(size_t)d * HEADS + hd];
  const float4 hv = *(const float4*)(H + (size_t)s * HID + lane * 4);
  float* op = agg + (size_t)d * HID + lane * 4;
  atomicAdd(op + 0, a * hv.x);
  atomicAdd(op + 1, a * hv.y);
  atomicAdd(op + 2, a * hv.z);
  atomicAdd(op + 3, a * hv.w);
}

// ------------- bias + ELU (in-place: agg1 -> h2) -------------
__global__ __launch_bounds__(256) void elu_kernel(float* __restrict__ agg,
                                                  const float* __restrict__ bias) {
  long i = (long)blockIdx.x * blockDim.x + threadIdx.x;
  if (i >= (long)NN * HID) return;
  float v = agg[i] + bias[i & (HID - 1)];
  agg[i] = (v > 0.f) ? v : (__expf(v) - 1.0f);
}

// ------------- layer-2 linear + attention coeffs -------------
__global__ __launch_bounds__(256) void lin2_kernel(const float* __restrict__ H2,
                                                   const float* __restrict__ W2,
                                                   const float* __restrict__ attS,
                                                   const float* __restrict__ attD,
                                                   float* __restrict__ olin,
                                                   float* __restrict__ aS,
                                                   float* __restrict__ aD,
                                                   float* __restrict__ m2,
                                                   float* __restrict__ dn2) {
  int n = blockIdx.x * blockDim.x + threadIdx.x;
  if (n >= NN) return;
  const float* hp = H2 + (size_t)n * HID;
  float o0 = 0.f, o1 = 0.f;
#pragma unroll 8
  for (int k = 0; k < HID; ++k) { float h = hp[k]; o0 += h * W2[k * 2]; o1 += h * W2[k * 2 + 1]; }
  olin[n * 2]     = o0;
  olin[n * 2 + 1] = o1;
  aS[n] = o0 * attS[0] + o1 * attS[1];
  aD[n] = o0 * attD[0] + o1 * attD[1];
  m2[n] = -INFINITY; dn2[n] = 0.0f;
}

// ------------- layer-2 edge passes (H=1) -------------
__global__ __launch_bounds__(256) void edge_a2(const int* __restrict__ ei,
                                               const float* __restrict__ aS,
                                               const float* __restrict__ aD,
                                               float* __restrict__ ebuf,
                                               float* __restrict__ m2) {
  long e = (long)blockIdx.x * blockDim.x + threadIdx.x;
  if (e >= ETOT) return;
  int s, d; edge_nodes(ei, e, s, d);
  float v = aS[s] + aD[d];
  v = (v < 0.f) ? NEG_SLOPE * v : v;
  ebuf[e] = v;
  atomicMaxF(&m2[d], v);
}

__global__ __launch_bounds__(256) void edge_b2(const int* __restrict__ ei,
                                               const float* __restrict__ m2,
                                               float* __restrict__ ebuf,
                                               float* __restrict__ dn2) {
  long e = (long)blockIdx.x * blockDim.x + threadIdx.x;
  if (e >= ETOT) return;
  int s, d; edge_nodes(ei, e, s, d);
  float ex = __expf(ebuf[e] - m2[d]);
  ebuf[e] = ex;
  atomicAdd(&dn2[d], ex);
}

__global__ __launch_bounds__(256) void edge_c2(const int* __restrict__ ei,
                                               const float* __restrict__ ebuf,
                                               const float* __restrict__ dn2,
                                               const float* __restrict__ olin,
                                               float* __restrict__ agg2) {
  long e = (long)blockIdx.x * blockDim.x + threadIdx.x;
  if (e >= ETOT) return;
  int s, d; edge_nodes(ei, e, s, d);
  float a = ebuf[e] / dn2[d];
  atomicAdd(&agg2[(size_t)d * 2 + 0], a * olin[(size_t)s * 2 + 0]);
  atomicAdd(&agg2[(size_t)d * 2 + 1], a * olin[(size_t)s * 2 + 1]);
}

// ------------- bias + log_softmax over 2 classes -------------
__global__ __launch_bounds__(256) void final_kernel(const float* __restrict__ agg2,
                                                    const float* __restrict__ bias2,
                                                    float* __restrict__ out) {
  int n = blockIdx.x * blockDim.x + threadIdx.x;
  if (n >= NN) return;
  float v0 = agg2[n * 2]     + bias2[0];
  float v1 = agg2[n * 2 + 1] + bias2[1];
  float mx = fmaxf(v0, v1);
  float lse = mx + __logf(__expf(v0 - mx) + __expf(v1 - mx));
  out[n * 2]     = v0 - lse;
  out[n * 2 + 1] = v1 - lse;
}

// =====================================================================
extern "C" void kernel_launch(void* const* d_in, const int* in_sizes, int n_in,
                              void* d_out, int out_size, void* d_ws, size_t ws_size,
                              hipStream_t stream) {
  (void)in_sizes; (void)n_in; (void)out_size; (void)ws_size;
  const float* x      = (const float*)d_in[0];
  const int*   ei     = (const int*)  d_in[1];
  const float* W1     = (const float*)d_in[2];
  const float* attS1  = (const float*)d_in[3];
  const float* attD1  = (const float*)d_in[4];
  const float* bias1  = (const float*)d_in[5];
  const float* W2     = (const float*)d_in[6];
  const float* attS2  = (const float*)d_in[7];
  const float* attD2  = (const float*)d_in[8];
  const float* bias2  = (const float*)d_in[9];
  float* out = (float*)d_out;

  // workspace carve-up (256B aligned)
  uintptr_t base = (uintptr_t)d_ws;
  size_t off = 0;
  auto carve = [&](size_t bytes) -> float* {
    float* p = (float*)(base + off);
    off = (off + bytes + 255) & ~(size_t)255;
    return p;
  };
  float* h1   = carve((size_t)NN * HID * 4);      // x @ W1
  float* agg1 = carve((size_t)NN * HID * 4);      // layer-1 aggregation -> h2 (in place)
  float* aS1  = carve((size_t)NN * HEADS * 4);
  float* aD1  = carve((size_t)NN * HEADS * 4);
  float* m1   = carve((size_t)NN * HEADS * 4);
  float* dn1  = carve((size_t)NN * HEADS * 4);
  float* ebuf = carve((size_t)ETOT * HEADS * 4);  // reused for layer-2 (needs ETOT*4)
  float* olin = carve((size_t)NN * NCLASS * 4);
  float* aS2  = carve((size_t)NN * 4);
  float* aD2  = carve((size_t)NN * 4);
  float* m2   = carve((size_t)NN * 4);
  float* dn2  = carve((size_t)NN * 4);
  float* agg2 = carve((size_t)NN * NCLASS * 4);

  const int B = 256;

  // zero accumulators (atomically updated each call)
  zero_kernel<<<2048, B, 0, stream>>>(agg1, (long)NN * HID);
  zero_kernel<<<256,  B, 0, stream>>>(agg2, (long)NN * NCLASS);

  // ---- layer 1 ----
  gemm1_wmma<<<(NTILES + 7) / 8, B, 0, stream>>>(x, W1, h1);
  alpha1_kernel<<<(NN * HEADS + B - 1) / B, B, 0, stream>>>(h1, attS1, attD1,
                                                            aS1, aD1, m1, dn1);
  {
    long t = (long)ETOT * HEADS;
    int g = (int)((t + B - 1) / B);
    edge_a1<<<g, B, 0, stream>>>(ei, aS1, aD1, ebuf, m1);
    edge_b1<<<g, B, 0, stream>>>(ei, m1, ebuf, dn1);
  }
  edge_c1<<<(ETOT + 7) / 8, B, 0, stream>>>(ei, ebuf, dn1, h1, agg1);
  elu_kernel<<<(int)(((long)NN * HID + B - 1) / B), B, 0, stream>>>(agg1, bias1);

  // ---- layer 2 ----
  lin2_kernel<<<(NN + B - 1) / B, B, 0, stream>>>(agg1, W2, attS2, attD2,
                                                  olin, aS2, aD2, m2, dn2);
  {
    int g = (ETOT + B - 1) / B;
    edge_a2<<<g, B, 0, stream>>>(ei, aS2, aD2, ebuf, m2);
    edge_b2<<<g, B, 0, stream>>>(ei, m2, ebuf, dn2);
    edge_c2<<<g, B, 0, stream>>>(ei, ebuf, dn2, olin, agg2);
  }
  final_kernel<<<(NN + B - 1) / B, B, 0, stream>>>(agg2, bias2, out);
}
